// ChimeraCognitive_12996571038043
// MI455X (gfx1250) — matrix operand order repaired and checked by reading, not compiled, not run on hardware
//
#include <hip/hip_runtime.h>
#include <hip/hip_bf16.h>
#include <math.h>

typedef __attribute__((ext_vector_type(16))) _Float16 v16h;
typedef __attribute__((ext_vector_type(8)))  _Float16 v8h;
typedef __attribute__((ext_vector_type(8)))  float    v8f;

#define LATENT   128
#define LEAK     0.1f
#define M_TILE   16      // batch rows per workgroup
#define T_CHUNK  128     // timesteps staged in LDS per chunk

// ---- hardware tanh (v_tanh_f32 on CDNA5) --------------------------------
__device__ __forceinline__ float fast_tanh(float x) {
#if __has_builtin(__builtin_amdgcn_tanhf)
    return __builtin_amdgcn_tanhf(x);
#elif __has_builtin(__builtin_amdgcn_tanh_f32)
    return __builtin_amdgcn_tanh_f32(x);
#else
    const float ax = __builtin_fabsf(x);
    const float e  = __builtin_amdgcn_exp2f(ax * -2.8853900817779268f);
    const float r  = (1.0f - e) * __builtin_amdgcn_rcpf(1.0f + e);
    return __builtin_copysignf(r, x);
#endif
}

// ---- 16-lane sum via DPP (no LDS traffic, co-executes with WMMA) --------
// Butterfly with xor masks {1,2,7,15}: quad_perm(1,0,3,2)=0xB1,
// quad_perm(2,3,0,1)=0x4E, ROW_HALF_MIRROR=0x141, ROW_MIRROR=0x140.
#if __has_builtin(__builtin_amdgcn_update_dpp)
template <int CTRL>
__device__ __forceinline__ float dpp_add(float v) {
    int sh = __builtin_amdgcn_update_dpp(0, __builtin_bit_cast(int, v),
                                         CTRL, 0xF, 0xF, true);
    return v + __builtin_bit_cast(float, sh);
}
__device__ __forceinline__ float red16(float s) {
    s = dpp_add<0xB1>(s);   // lane ^ 1
    s = dpp_add<0x4E>(s);   // lane ^ 2
    s = dpp_add<0x141>(s);  // lane ^ 7  (half mirror)
    s = dpp_add<0x140>(s);  // lane ^ 15 (row mirror)
    return s;
}
#else
__device__ __forceinline__ float red16(float s) {
    s += __shfl_xor(s, 1);
    s += __shfl_xor(s, 2);
    s += __shfl_xor(s, 4);
    s += __shfl_xor(s, 8);
    return s;
}
#endif

// ---- split workgroup barrier: overlap work between signal and wait ------
#if __has_builtin(__builtin_amdgcn_s_barrier_signal) && \
    __has_builtin(__builtin_amdgcn_s_barrier_wait)
#define BAR_SIGNAL()                                          \
    do {                                                      \
        __builtin_amdgcn_fence(__ATOMIC_RELEASE, "workgroup");\
        __builtin_amdgcn_s_barrier_signal(-1);                \
    } while (0)
#define BAR_WAIT()                                            \
    do {                                                      \
        __builtin_amdgcn_s_barrier_wait(-1);                  \
        __builtin_amdgcn_fence(__ATOMIC_ACQUIRE, "workgroup");\
    } while (0)
#else
#define BAR_SIGNAL() do {} while (0)
#define BAR_WAIT()   __syncthreads()
#endif

// One workgroup = 256 threads = 8 waves. Wave w owns output columns
// n in [16w, 16w+16). Lane layout follows the documented C/D 16x16 f32
// layout: VGPR v -> row m = v + 8*half, column n = 16w + (lane&15).
__global__ __launch_bounds__(256)
void esn_wmma_kernel(const float* __restrict__ x,      // [B, S, 2]
                     const float* __restrict__ W_res,  // [128, 128]
                     const float* __restrict__ W_in,   // [2, 128]
                     const float* __restrict__ bias,   // [128]
                     const float* __restrict__ W_out,  // [128, 1]
                     const float* __restrict__ b_out,  // [1]
                     float* __restrict__ out,          // [B, S, 1]
                     int S)
{
    __shared__ _Float16 h_lds[2][M_TILE * LATENT];     // double-buffered f16 h
    __shared__ float    x_lds[M_TILE * T_CHUNK * 2];   // [m][tt][c]
    __shared__ float    o_lds[M_TILE * T_CHUNK];       // [m][tt]

    const int tid    = threadIdx.x;
    const int wave   = tid >> 5;
    const int lane   = tid & 31;
    const int halfId = lane >> 4;
    const int l16    = lane & 15;
    const int n      = wave * 16 + l16;
    const int b0     = blockIdx.x * M_TILE;

    // ---- per-lane constants ----
    const float w_in0 = W_in[n];
    const float w_in1 = W_in[LATENT + n];
    const float bn    = bias[n];
    const float wout  = W_out[n];
    const float bout  = b_out[0];

    // ---- W_res -> persistent f16 B-fragments (reused 2048x) ----
    v16h bfrag[4];
#pragma unroll
    for (int c = 0; c < 4; ++c) {
#pragma unroll
        for (int j = 0; j < 16; ++j) {
            const int k = 32 * c + (j < 8 ? 8 * halfId + j : 16 + 8 * halfId + (j - 8));
            bfrag[c][j] = (_Float16)W_res[k * LATENT + n];
        }
    }

    // ---- init h ----
    float hacc[8];
#pragma unroll
    for (int v = 0; v < 8; ++v) hacc[v] = 0.0f;
    for (int i = tid; i < M_TILE * LATENT; i += 256)
        h_lds[0][i] = (_Float16)0.0f;
    __syncthreads();

    int cur = 0;
    for (int t0 = 0; t0 < S; t0 += T_CHUNK) {
        // ---- stage x chunk, init output accumulator ----
        {
            const int m    = tid >> 4;
            const int part = tid & 15;
            const float* src = x + ((size_t)(b0 + m) * S + t0) * 2 + part * 16;
            float* dst = &x_lds[m * T_CHUNK * 2 + part * 16];
#pragma unroll
            for (int q = 0; q < 16; q += 4)
                *(float4*)(dst + q) = *(const float4*)(src + q);
            if (t0 + T_CHUNK < S)
                __builtin_prefetch(src + T_CHUNK * 2, 0, 0);
            float* od = &o_lds[m * T_CHUNK + part * 8];
#pragma unroll
            for (int q = 0; q < 8; ++q) od[q] = bout;
        }
        __syncthreads();

        // ---- recurrence: split barrier per step ----
        for (int tt = 0; tt < T_CHUNK; ++tt) {
            const int nxt = cur ^ 1;

            // A-fragments of h (documented A 16x32 f16 layout; 2x b128 per chunk)
            const _Float16* hrow = &h_lds[cur][l16 * LATENT];
            v16h afrag[4];
#pragma unroll
            for (int c = 0; c < 4; ++c) {
                const v8h g0 = *(const v8h*)(hrow + 32 * c + 8 * halfId);
                const v8h g1 = *(const v8h*)(hrow + 32 * c + 16 + 8 * halfId);
                v16h a;
#pragma unroll
                for (int j = 0; j < 8; ++j) { a[j] = g0[j]; a[8 + j] = g1[j]; }
                afrag[c] = a;
            }

            // C operand = bias + u_t @ W_in
            v8f d;
#pragma unroll
            for (int v = 0; v < 8; ++v) {
                const int m = v + 8 * halfId;
                const float2 u = *(const float2*)&x_lds[m * T_CHUNK * 2 + tt * 2];
                d[v] = bn + u.x * w_in0 + u.y * w_in1;
            }

            // drive = h @ W_res + C : 4 chained WMMAs over K=128
            d = __builtin_amdgcn_wmma_f32_16x16x32_f16(false, afrag[0], false, bfrag[0], (short)0, d, false, false);
            d = __builtin_amdgcn_wmma_f32_16x16x32_f16(false, afrag[1], false, bfrag[1], (short)0, d, false, false);
            d = __builtin_amdgcn_wmma_f32_16x16x32_f16(false, afrag[2], false, bfrag[2], (short)0, d, false, false);
            d = __builtin_amdgcn_wmma_f32_16x16x32_f16(false, afrag[3], false, bfrag[3], (short)0, d, false, false);

            // leaky update (v_tanh_f32 + pk fma), publish f16 h
#pragma unroll
            for (int v = 0; v < 8; ++v) {
                const float hn = fast_tanh(d[v]);
                hacc[v] = (1.0f - LEAK) * hacc[v] + LEAK * hn;
                h_lds[nxt][(v + 8 * halfId) * LATENT + n] = (_Float16)hacc[v];
            }

            // h published: signal now, overlap projection with barrier latency
            BAR_SIGNAL();

            // fused output projection: DPP butterfly, one ds_add per row
            float p[8];
#pragma unroll
            for (int v = 0; v < 8; ++v)
                p[v] = red16(hacc[v] * wout);
            if (l16 == 0) {
#pragma unroll
                for (int v = 0; v < 8; ++v)
                    atomicAdd(&o_lds[(v + 8 * halfId) * T_CHUNK + tt], p[v]);
            }

            BAR_WAIT();
            cur = nxt;
        }

        // last step's o_lds atomics land after its signal -> full barrier here
        __syncthreads();

        // ---- bulk store output chunk (coalesced per batch row) ----
        {
            const int m    = tid >> 4;
            const int part = tid & 15;
            float* dst = out + (size_t)(b0 + m) * S + t0 + part * 8;
            const float* sl = &o_lds[m * T_CHUNK + part * 8];
            *(float4*)(dst + 0) = *(const float4*)(sl + 0);
            *(float4*)(dst + 4) = *(const float4*)(sl + 4);
        }
        __syncthreads();               // before next chunk reuses x_lds/o_lds
    }
}

extern "C" void kernel_launch(void* const* d_in, const int* in_sizes, int n_in,
                              void* d_out, int out_size, void* d_ws, size_t ws_size,
                              hipStream_t stream) {
    const float* x     = (const float*)d_in[0];
    const float* W_res = (const float*)d_in[1];
    const float* W_in  = (const float*)d_in[2];
    const float* bias  = (const float*)d_in[3];
    const float* W_out = (const float*)d_in[4];
    const float* b_out = (const float*)d_in[5];
    float* out = (float*)d_out;

    const int B = 512;
    const int S = out_size / B;   // 2048

    dim3 grid(B / M_TILE);        // 32 workgroups
    dim3 block(256);              // 8 waves (wave32)
    esn_wmma_kernel<<<grid, block, 0, stream>>>(x, W_res, W_in, bias, W_out,
                                                b_out, out, S);
}
